// ModifiedMultiheadAttention_15779709846136
// MI455X (gfx1250) — matrix operand order repaired and checked by reading, not compiled, not run on hardware
//
#include <hip/hip_runtime.h>
#include <hip/hip_bf16.h>

// Problem constants
#define BB   8
#define SS   1024
#define IND  512
#define EMB  512
#define NH   8
#define HD   64
#define QKVN 1536   // 3*EMB
#define LNN  ((float)(SS * EMB))

typedef __attribute__((ext_vector_type(16))) _Float16 v16h;
typedef __attribute__((ext_vector_type(8)))  _Float16 v8h;
typedef __attribute__((ext_vector_type(4)))  _Float16 v4h;
typedef __attribute__((ext_vector_type(8)))  float    v8f;
typedef __attribute__((ext_vector_type(4)))  float    v4f;

// ---------------- WMMA helpers (CDNA5 wave32) ----------------

__device__ inline v8f wmma_f16(v16h a, v16h b, v8f c) {
  // v_wmma_f32_16x16x32_f16 : D = A(16x32 f16) * B(32x16 f16) + C(16x16 f32)
  return __builtin_amdgcn_wmma_f32_16x16x32_f16(
      /*neg_a=*/false, a, /*neg_b=*/false, b,
      /*c_mod=*/(short)0, c, /*reuse_a=*/false, /*reuse_b=*/false);
}

// A fragment, 16x32 f16, from LDS tile stored row-major [row][k], row stride ld.
// Per ISA: lanes0-15 row=lane, elems0-7 = K0-7, elems8-15 = K16-23;
//          lanes16-31 row=lane-16, elems0-7 = K8-15, elems8-15 = K24-31.
__device__ inline v16h load_frag_a(const _Float16* base, int ld, int kofs) {
  int lane = threadIdx.x & 31;
  int row  = lane & 15;
  int k0   = (lane >> 4) * 8;
  const _Float16* p = base + row * ld + kofs + k0;
  v8h lo = *(const v8h*)(p);
  v8h hi = *(const v8h*)(p + 16);
  v16h r;
#pragma unroll
  for (int i = 0; i < 8; ++i) { r[i] = lo[i]; r[i + 8] = hi[i]; }
  return r;
}

// B fragment, 32x16 f16, from LDS tile stored [n][k] (column of B contiguous
// over K), row stride ld. lanes0-15: n=lane, K0-7,K16-23; lanes16-31: n=lane-16, K8-15,K24-31.
__device__ inline v16h load_frag_b(const _Float16* base, int ld, int kofs) {
  int lane = threadIdx.x & 31;
  int n    = lane & 15;
  int k0   = (lane >> 4) * 8;
  const _Float16* p = base + n * ld + kofs + k0;
  v8h lo = *(const v8h*)(p);
  v8h hi = *(const v8h*)(p + 16);
  v16h r;
#pragma unroll
  for (int i = 0; i < 8; ++i) { r[i] = lo[i]; r[i + 8] = hi[i]; }
  return r;
}

// cvt + store one 4-float chunk as 4 halves (8B LDS store)
__device__ inline void cvt_store4(_Float16* dst, const float* src) {
  v4f f = *(const v4f*)src;
  v4h h;
#pragma unroll
  for (int j = 0; j < 4; ++j) h[j] = (_Float16)f[j];
  *(v4h*)dst = h;
}

// ---------------- Kernel 0: zero LN accumulators ----------------

__global__ void init_sums_kernel(float* sums) {
  if (threadIdx.x < 2 * BB) sums[threadIdx.x] = 0.0f;
}

// ---------------- Kernel 1: QKV projection  qkv = x @ w_qkv^T ----------------
// M = B*S = 8192, N = 1536, K = 512.
// Block: 128 thr (4 waves), block tile 128x64; each wave computes 32x64
// (2 A-frags x 4 B-frags -> 8 wmma per 32-wide k-step).

#define GLDA 40
#define GLDB 40

__global__ __launch_bounds__(128) void qkv_gemm_kernel(
    const float* __restrict__ x, const float* __restrict__ w,
    _Float16* __restrict__ qkv) {
  __shared__ alignas(16) _Float16 At[128 * GLDA];
  __shared__ alignas(16) _Float16 Bt[64 * GLDB];
  const int mt = blockIdx.x;        // 64 tiles of 128 rows
  const int nt = blockIdx.y;        // 24 tiles of 64 cols
  const int tid = threadIdx.x, wave = tid >> 5, lane = tid & 31;

  v8f acc[2][4] = {};
  for (int kk = 0; kk < IND; kk += 32) {
    __syncthreads();
    // prefetch next k-tile (one cacheline per row per thread)
    if (kk + 32 < IND) {
      __builtin_prefetch(&x[(size_t)(mt * 128 + tid) * IND + kk + 32], 0, 1);
      if (tid < 64)
        __builtin_prefetch(&w[(size_t)(nt * 64 + tid) * IND + kk + 32], 0, 1);
    }
    // A: 128x32 f32 -> f16, vectorized: 1024 chunks of 4
#pragma unroll
    for (int idx = tid; idx < 128 * 8; idx += 128) {
      int r = idx >> 3, c4 = (idx & 7) * 4;
      cvt_store4(&At[r * GLDA + c4], &x[(size_t)(mt * 128 + r) * IND + kk + c4]);
    }
    // B: 64x32 f32 -> f16, 512 chunks of 4
#pragma unroll
    for (int idx = tid; idx < 64 * 8; idx += 128) {
      int n = idx >> 3, c4 = (idx & 7) * 4;
      cvt_store4(&Bt[n * GLDB + c4], &w[(size_t)(nt * 64 + n) * IND + kk + c4]);
    }
    __syncthreads();
    v16h a0 = load_frag_a(At + (wave * 32 + 0)  * GLDA, GLDA, 0);
    v16h a1 = load_frag_a(At + (wave * 32 + 16) * GLDA, GLDA, 0);
#pragma unroll
    for (int ns = 0; ns < 4; ++ns) {
      v16h b = load_frag_b(Bt + ns * 16 * GLDB, GLDB, 0);
      acc[0][ns] = wmma_f16(a0, b, acc[0][ns]);
      acc[1][ns] = wmma_f16(a1, b, acc[1][ns]);
    }
  }
  const int half = lane >> 4, col0 = lane & 15;
#pragma unroll
  for (int ms = 0; ms < 2; ++ms)
#pragma unroll
    for (int ns = 0; ns < 4; ++ns)
#pragma unroll
      for (int r = 0; r < 8; ++r) {
        int row = mt * 128 + wave * 32 + ms * 16 + r + half * 8;
        int col = nt * 64 + ns * 16 + col0;
        qkv[(size_t)row * QKVN + col] = (_Float16)acc[ms][ns][r];
      }
}

// ---------------- Kernel 2: flash attention per (b,h,128-query block) --------
// 8 waves; wave owns 16 query rows. K-blocks of 32 keys, online softmax.

#define LDQ  72
#define LDKT 72
#define LDVT 40
#define LDP  40

__global__ __launch_bounds__(256) void attn_kernel(
    const _Float16* __restrict__ qkv, _Float16* __restrict__ vals) {
  __shared__ alignas(16) _Float16 Qs[128 * LDQ];
  __shared__ alignas(16) _Float16 Ks[32 * LDKT];
  __shared__ alignas(16) _Float16 Vs[64 * LDVT];       // transposed: [d][key]
  __shared__ alignas(16) _Float16 Ps[8 * 16 * LDP];    // per-wave P tile
  const int qb = blockIdx.x, h = blockIdx.y, b = blockIdx.z;
  const int tid = threadIdx.x, wave = tid >> 5, lane = tid & 31;
  const int half = lane >> 4;
  const _Float16* qkvb = qkv + (size_t)b * SS * QKVN;

  // stage Q block 128x64 into LDS (v8h chunks), then pull A-fragments
#pragma unroll
  for (int idx = tid; idx < 128 * 8; idx += 256) {
    int r = idx >> 3, c8 = (idx & 7) * 8;
    *(v8h*)&Qs[r * LDQ + c8] =
        *(const v8h*)&qkvb[(size_t)(qb * 128 + r) * QKVN + h * 192 + c8];
  }
  __syncthreads();
  v16h aq0 = load_frag_a(Qs + wave * 16 * LDQ, LDQ, 0);
  v16h aq1 = load_frag_a(Qs + wave * 16 * LDQ, LDQ, 32);

  v8f o[4] = {};
  float m[8], l[8];
#pragma unroll
  for (int r = 0; r < 8; ++r) { m[r] = -1e30f; l[r] = 0.0f; }

  const int kx = tid >> 3, c8 = (tid & 7) * 8;   // staging coords (1 chunk/thread)
  for (int kb = 0; kb < SS / 32; ++kb) {
    __syncthreads();
    // K tile [key][d]: 32x64 halves = 256 v8h chunks, one per thread
    *(v8h*)&Ks[kx * LDKT + c8] =
        *(const v8h*)&qkvb[(size_t)(kb * 32 + kx) * QKVN + h * 192 + 64 + c8];
    // V tile transposed [d][key]: vector global load, scalar LDS scatter
    {
      v8h vv = *(const v8h*)&qkvb[(size_t)(kb * 32 + kx) * QKVN + h * 192 + 128 + c8];
#pragma unroll
      for (int j = 0; j < 8; ++j) Vs[(c8 + j) * LDVT + kx] = vv[j];
    }
    __syncthreads();

    // logits: two 16x16 key sub-tiles, K-dim = 64 (two wmma each)
    v8f s0 = {}, s1 = {};
    s0 = wmma_f16(aq0, load_frag_b(Ks, LDKT, 0), s0);
    s0 = wmma_f16(aq1, load_frag_b(Ks, LDKT, 32), s0);
    s1 = wmma_f16(aq0, load_frag_b(Ks + 16 * LDKT, LDKT, 0), s1);
    s1 = wmma_f16(aq1, load_frag_b(Ks + 16 * LDKT, LDKT, 32), s1);

    const float scale = 0.125f;   // 1/sqrt(64)
    float alpha[8];
#pragma unroll
    for (int r = 0; r < 8; ++r) {
      float v0 = s0[r] * scale, v1 = s1[r] * scale;
      float rm = fmaxf(v0, v1);
      rm = fmaxf(rm, __shfl_xor(rm, 1, 32));
      rm = fmaxf(rm, __shfl_xor(rm, 2, 32));
      rm = fmaxf(rm, __shfl_xor(rm, 4, 32));
      rm = fmaxf(rm, __shfl_xor(rm, 8, 32));
      float mn = fmaxf(m[r], rm);
      alpha[r] = __expf(m[r] - mn);
      m[r] = mn;
      float p0 = __expf(v0 - mn), p1 = __expf(v1 - mn);
      float rs = p0 + p1;
      rs += __shfl_xor(rs, 1, 32);
      rs += __shfl_xor(rs, 2, 32);
      rs += __shfl_xor(rs, 4, 32);
      rs += __shfl_xor(rs, 8, 32);
      l[r] = l[r] * alpha[r] + rs;
      // scatter P into per-wave LDS tile [row][key32] for the A-fragment
      _Float16* pw = Ps + wave * 16 * LDP;
      pw[(r + half * 8) * LDP + (lane & 15)]      = (_Float16)p0;
      pw[(r + half * 8) * LDP + 16 + (lane & 15)] = (_Float16)p1;
    }
#pragma unroll
    for (int t = 0; t < 4; ++t)
#pragma unroll
      for (int r = 0; r < 8; ++r) o[t][r] *= alpha[r];

    __syncthreads();   // make P tile visible before fragment reload
    v16h ap = load_frag_a(Ps + wave * 16 * LDP, LDP, 0);
#pragma unroll
    for (int t = 0; t < 4; ++t)
      o[t] = wmma_f16(ap, load_frag_b(Vs + t * 16 * LDVT, LDVT, 0), o[t]);
  }

  // normalize and write vals[b, s, h*64 + d]  (f16)
#pragma unroll
  for (int t = 0; t < 4; ++t)
#pragma unroll
    for (int r = 0; r < 8; ++r) {
      int row = qb * 128 + wave * 16 + r + half * 8;
      int col = h * HD + t * 16 + (lane & 15);
      vals[((size_t)b * SS + row) * EMB + col] = (_Float16)(o[t][r] / l[r]);
    }
}

// ---------------- Kernel 3: O-proj  o = vals @ w_o^T  (+ LN partial sums) ----
// Same 128x64 block tile / 32x64 wave tile as the QKV GEMM.

__global__ __launch_bounds__(128) void oproj_kernel(
    const _Float16* __restrict__ vals, const float* __restrict__ wo,
    float* __restrict__ out, float* __restrict__ sums) {
  __shared__ alignas(16) _Float16 At[128 * GLDA];
  __shared__ alignas(16) _Float16 Bt[64 * GLDB];
  const int mt = blockIdx.x;   // 64 tiles of 128 rows (8 per batch)
  const int nt = blockIdx.y;   // 8 tiles of 64 cols
  const int tid = threadIdx.x, wave = tid >> 5, lane = tid & 31;

  v8f acc[2][4] = {};
  for (int kk = 0; kk < EMB; kk += 32) {
    __syncthreads();
    if (kk + 32 < EMB) {
      __builtin_prefetch(&vals[(size_t)(mt * 128 + tid) * EMB + kk + 32], 0, 1);
      if (tid < 64)
        __builtin_prefetch(&wo[(size_t)(nt * 64 + tid) * EMB + kk + 32], 0, 1);
    }
    // A: 128x32 f16 copy, v8h chunks (512 total)
#pragma unroll
    for (int idx = tid; idx < 128 * 4; idx += 128) {
      int r = idx >> 2, c8 = (idx & 3) * 8;
      *(v8h*)&At[r * GLDA + c8] =
          *(const v8h*)&vals[(size_t)(mt * 128 + r) * EMB + kk + c8];
    }
    // B: 64x32 f32 -> f16, 512 chunks of 4
#pragma unroll
    for (int idx = tid; idx < 64 * 8; idx += 128) {
      int n = idx >> 3, c4 = (idx & 7) * 4;
      cvt_store4(&Bt[n * GLDB + c4], &wo[(size_t)(nt * 64 + n) * EMB + kk + c4]);
    }
    __syncthreads();
    v16h a0 = load_frag_a(At + (wave * 32 + 0)  * GLDA, GLDA, 0);
    v16h a1 = load_frag_a(At + (wave * 32 + 16) * GLDA, GLDA, 0);
#pragma unroll
    for (int ns = 0; ns < 4; ++ns) {
      v16h b = load_frag_b(Bt + ns * 16 * GLDB, GLDB, 0);
      acc[0][ns] = wmma_f16(a0, b, acc[0][ns]);
      acc[1][ns] = wmma_f16(a1, b, acc[1][ns]);
    }
  }
  const int half = lane >> 4, col0 = lane & 15;
  float lsum = 0.0f, lsq = 0.0f;
#pragma unroll
  for (int ms = 0; ms < 2; ++ms)
#pragma unroll
    for (int ns = 0; ns < 4; ++ns)
#pragma unroll
      for (int r = 0; r < 8; ++r) {
        int row = mt * 128 + wave * 32 + ms * 16 + r + half * 8;
        int col = nt * 64 + ns * 16 + col0;
        float v = acc[ms][ns][r];
        out[(size_t)row * EMB + col] = v;
        lsum += v;
        lsq  += v * v;
      }
  // wave reduction (all 32 lanes hold disjoint outputs)
#pragma unroll
  for (int msk = 16; msk; msk >>= 1) {
    lsum += __shfl_xor(lsum, msk, 32);
    lsq  += __shfl_xor(lsq,  msk, 32);
  }
  const int bidx = mt >> 3;   // 8 m-tiles (of 128 rows) per batch
  if (lane == 0) {
    atomicAdd(&sums[bidx * 2 + 0], lsum);
    atomicAdd(&sums[bidx * 2 + 1], lsq);
  }
}

// ---------------- Kernel 4: LayerNorm finalize over [S,E] per batch ----------

__global__ void ln_kernel(float* __restrict__ out, const float* __restrict__ sums) {
  size_t i = (size_t)blockIdx.x * blockDim.x + threadIdx.x;
  if (i >= (size_t)BB * SS * EMB) return;
  int b = (int)(i / ((size_t)SS * EMB));
  float mu  = sums[b * 2 + 0] / LNN;
  float var = sums[b * 2 + 1] / LNN - mu * mu;
  float sc  = rsqrtf(var + 1e-5f);
  out[i] = (out[i] - mu) * sc;
}

// ---------------- host-side launch ----------------

extern "C" void kernel_launch(void* const* d_in, const int* in_sizes, int n_in,
                              void* d_out, int out_size, void* d_ws, size_t ws_size,
                              hipStream_t stream) {
  const float* x     = (const float*)d_in[0];   // [8,1024,512]
  const float* w_qkv = (const float*)d_in[1];   // [1536,512]
  const float* w_o   = (const float*)d_in[2];   // [512,512]
  float* out = (float*)d_out;                   // [8,1024,512]

  char* ws = (char*)d_ws;
  _Float16* qkv  = (_Float16*)ws;                                  // 8192*1536 f16 = 24 MB
  _Float16* vals = (_Float16*)(ws + (size_t)BB * SS * QKVN * 2);   // 8192*512 f16 = 8 MB
  float*    sums = (float*)(ws + (size_t)BB * SS * QKVN * 2
                               + (size_t)BB * SS * EMB * 2);       // 16 floats

  init_sums_kernel<<<1, 32, 0, stream>>>(sums);
  qkv_gemm_kernel<<<dim3((BB * SS) / 128, QKVN / 64), 128, 0, stream>>>(x, w_qkv, qkv);
  attn_kernel<<<dim3(SS / 128, NH, BB), 256, 0, stream>>>(qkv, vals);
  oproj_kernel<<<dim3((BB * SS) / 128, EMB / 64), 128, 0, stream>>>(vals, w_o, out, sums);
  size_t total = (size_t)BB * SS * EMB;
  ln_kernel<<<(unsigned)((total + 255) / 256), 256, 0, stream>>>(out, sums);
}
